// GPT_3272765079741
// MI455X (gfx1250) — matrix operand order repaired and checked by reading, not compiled, not run on hardware
//
#include <hip/hip_runtime.h>
#include <math.h>

#define C_DIM 1024
#define T_DIM 1024
#define B_DIM 2
#define H_DIM 8
#define D_DIM 128
#define L_DIM 8
#define M_ROWS (B_DIM * T_DIM)  // 2048 token rows

typedef __bf16 bf16_t;
typedef bf16_t v16bf __attribute__((ext_vector_type(16)));
typedef float  v8f   __attribute__((ext_vector_type(8)));

union Frag {
    v16bf v;
    uint4 q[2];
};

__device__ __forceinline__ unsigned short f2bf(float f) {
    unsigned int u = __float_as_uint(f);
    u += 0x7FFFu + ((u >> 16) & 1u);   // round-to-nearest-even
    return (unsigned short)(u >> 16);
}

// ---------------------------------------------------------------------------
// CDNA5 async global->LDS copy (GLOBAL_LOAD_ASYNC_TO_LDS_B128, ASYNCcnt).
// ldsAddr: DS byte address (low 32 bits of the flat shared-pointer; the LDS
// aperture maps addr[31:0] straight to LDS_ADDR per ISA 10.2).
// Each issue moves 16 bytes per lane directly into LDS.
// ---------------------------------------------------------------------------
__device__ __forceinline__ void async_copy_b128(unsigned ldsAddr,
                                                const unsigned short* gptr) {
    asm volatile("global_load_async_to_lds_b128 %0, %1, off"
                 :: "v"(ldsAddr), "v"(gptr)
                 : "memory");
}

__device__ __forceinline__ void wait_async_all() {
#if __has_builtin(__builtin_amdgcn_s_wait_asynccnt)
    __builtin_amdgcn_s_wait_asynccnt(0);
#else
    asm volatile("s_wait_asynccnt 0x0" ::: "memory");
#endif
}

// ---------------------------------------------------------------------------
// fp32 -> bf16 weight conversion (grid-stride)
// ---------------------------------------------------------------------------
__global__ __launch_bounds__(256)
void f32_to_bf16_kernel(const float* __restrict__ src,
                        unsigned short* __restrict__ dst, size_t n) {
    size_t i = (size_t)blockIdx.x * blockDim.x + threadIdx.x;
    size_t stride = (size_t)gridDim.x * blockDim.x;
    for (; i < n; i += stride) dst[i] = f2bf(src[i]);
}

// ---------------------------------------------------------------------------
// h = x + wpe   (fp32 residual stream)
// ---------------------------------------------------------------------------
__global__ __launch_bounds__(256)
void embed_kernel(const float* __restrict__ x, const float* __restrict__ wpe,
                  float* __restrict__ h) {
    size_t i = (size_t)blockIdx.x * 256 + threadIdx.x;
    if (i < (size_t)M_ROWS * C_DIM) {
        size_t t = (i / C_DIM) % T_DIM;
        h[i] = x[i] + wpe[t * C_DIM + (i % C_DIM)];
    }
}

// ---------------------------------------------------------------------------
// LayerNorm over rows of [M_ROWS, C].  BF16OUT: emit bf16 activations.
// ---------------------------------------------------------------------------
template <bool BF16OUT>
__global__ __launch_bounds__(256)
void ln_kernel(const float* __restrict__ x, const float* __restrict__ w,
               const float* __restrict__ bia, void* __restrict__ outv) {
    __shared__ float rs[256];
    __shared__ float rq[256];
    const int row = blockIdx.x;
    const int tid = threadIdx.x;
    const float* xr = x + (size_t)row * C_DIM;
    float s = 0.f, q = 0.f;
    #pragma unroll
    for (int i = 0; i < C_DIM / 256; ++i) {
        float v = xr[tid + i * 256];
        s += v;
        q += v * v;
    }
    rs[tid] = s;
    rq[tid] = q;
    __syncthreads();
    for (int off = 128; off > 0; off >>= 1) {
        if (tid < off) {
            rs[tid] += rs[tid + off];
            rq[tid] += rq[tid + off];
        }
        __syncthreads();
    }
    float mean = rs[0] * (1.0f / C_DIM);
    float var  = rq[0] * (1.0f / C_DIM) - mean * mean;
    float rstd = rsqrtf(var + 1e-5f);
    #pragma unroll
    for (int i = 0; i < C_DIM / 256; ++i) {
        int c = tid + i * 256;
        float v = (xr[c] - mean) * rstd * w[c] + bia[c];
        if (BF16OUT)
            ((unsigned short*)outv)[(size_t)row * C_DIM + c] = f2bf(v);
        else
            ((float*)outv)[(size_t)row * C_DIM + c] = v;
    }
}

// ---------------------------------------------------------------------------
// bf16 WMMA GEMM: C[m,n] = sum_k A[m,k] * W[n,k] + bias[n]
//   A: [M,K] bf16 row-major, W: [N,K] bf16 row-major (x @ W^T form)
// EPI: 0 = bf16 out; 1 = tanh-GELU then bf16 out; 2 = +residual, f32 out
// Block tile 128x128, BK=32, 8 waves (2x4), wave tile 64x32 -> acc[4][2].
// Double-buffered LDS; the next K-slice streams in via async global->LDS
// copies (ASYNCcnt) while the 8 WMMAs of the current slice execute.
// ---------------------------------------------------------------------------
template <int EPI>
__global__ __launch_bounds__(256)
void gemm_bf16(const unsigned short* __restrict__ A,
               const unsigned short* __restrict__ W,
               const float* __restrict__ bias,
               void* __restrict__ outv,
               const float* __restrict__ res,
               int M, int N, int K) {
    __shared__ __align__(16) unsigned short As[2][128][40];
    __shared__ __align__(16) unsigned short Ws[2][128][40];

    const int tid  = threadIdx.x;
    const int lane = tid & 31;
    const int wid  = tid >> 5;
    const int wr   = wid & 1;    // 2 wave rows * 64
    const int wc   = wid >> 1;   // 4 wave cols * 32
    const int ln16 = lane & 15;
    const int half = lane >> 4;

    const int m0 = blockIdx.y * 128;
    const int n0 = blockIdx.x * 128;

    const v8f vz = {0.f, 0.f, 0.f, 0.f, 0.f, 0.f, 0.f, 0.f};
    v8f acc[4][2];
    #pragma unroll
    for (int i = 0; i < 4; ++i)
        #pragma unroll
        for (int j = 0; j < 2; ++j) acc[i][j] = vz;

    // staging: each thread owns one 32-byte half-row of A and of W
    const int srow = tid >> 1;
    const int scol = (tid & 1) * 16;
    const unsigned short* aPtr = A + (size_t)(m0 + srow) * K + scol;
    const unsigned short* wPtr = W + (size_t)(n0 + srow) * K + scol;

    unsigned aLds[2], wLds[2];
    aLds[0] = (unsigned)(size_t)&As[0][srow][scol];
    aLds[1] = (unsigned)(size_t)&As[1][srow][scol];
    wLds[0] = (unsigned)(size_t)&Ws[0][srow][scol];
    wLds[1] = (unsigned)(size_t)&Ws[1][srow][scol];

    auto stageAsync = [&](int k0, int bufSel) {
        async_copy_b128(aLds[bufSel],      aPtr + k0);
        async_copy_b128(aLds[bufSel] + 16, aPtr + k0 + 8);
        async_copy_b128(wLds[bufSel],      wPtr + k0);
        async_copy_b128(wLds[bufSel] + 16, wPtr + k0 + 8);
    };

    stageAsync(0, 0);
    wait_async_all();
    __syncthreads();

    int buf = 0;
    for (int k0 = 0; k0 < K; k0 += 32) {
        const bool more = (k0 + 32) < K;
        if (more) {
            stageAsync(k0 + 32, buf ^ 1);       // DMA overlaps WMMAs below
            if (k0 + 64 < K) {
                __builtin_prefetch(aPtr + k0 + 64, 0, 0);
                __builtin_prefetch(wPtr + k0 + 64, 0, 0);
            }
        }

        Frag a[4], b[2];
        const int base = half ? 8 : 0;
        #pragma unroll
        for (int i = 0; i < 4; ++i) {
            const unsigned short* ap = &As[buf][wr * 64 + i * 16 + ln16][0];
            a[i].q[0] = *reinterpret_cast<const uint4*>(ap + base);
            a[i].q[1] = *reinterpret_cast<const uint4*>(ap + base + 16);
        }
        const int kb = half ? 8 : 0;
        #pragma unroll
        for (int j = 0; j < 2; ++j) {
            const unsigned short* wp = &Ws[buf][wc * 32 + j * 16 + ln16][kb];
            b[j].q[0] = *reinterpret_cast<const uint4*>(wp);
            b[j].q[1] = *reinterpret_cast<const uint4*>(wp + 16);
        }
        #pragma unroll
        for (int i = 0; i < 4; ++i)
            #pragma unroll
            for (int j = 0; j < 2; ++j)
                acc[i][j] = __builtin_amdgcn_wmma_f32_16x16x32_bf16(
                    false, a[i].v, false, b[j].v, (short)0, acc[i][j], false, false);

        if (more) {
            wait_async_all();
            __syncthreads();
            buf ^= 1;
        }
    }

    // Epilogue (C/D layout: vgpr r -> row r (+8 for lanes 16-31), col = lane%16)
    #pragma unroll
    for (int i = 0; i < 4; ++i) {
        #pragma unroll
        for (int j = 0; j < 2; ++j) {
            #pragma unroll
            for (int r = 0; r < 8; ++r) {
                int row = m0 + wr * 64 + i * 16 + r + half * 8;
                int col = n0 + wc * 32 + j * 16 + ln16;
                float val = acc[i][j][r] + bias[col];
                if (EPI == 1) {
                    float xg = val;
                    val = 0.5f * xg *
                          (1.0f + tanhf(0.7978845608028654f *
                                        (xg + 0.044715f * xg * xg * xg)));
                }
                size_t o = (size_t)row * N + col;
                if (EPI == 2) {
                    ((float*)outv)[o] = val + res[o];
                } else {
                    ((unsigned short*)outv)[o] = f2bf(val);
                }
            }
        }
    }
}

// ---------------------------------------------------------------------------
// Transpose the V third of qkv into vT[(b*H+h)*D + d][t] so attention's
// P@V B-fragments become contiguous 16-byte loads.  32x32 LDS tile transpose.
// ---------------------------------------------------------------------------
__global__ __launch_bounds__(256)
void v_transpose_kernel(const unsigned short* __restrict__ qkv,
                        unsigned short* __restrict__ vT) {
    __shared__ unsigned short tile[32][33];
    const int tb  = blockIdx.x;          // t tile (T/32)
    const int vrb = blockIdx.y;          // vr tile ((B*H*D)/32)
    const int vrbase = vrb * 32;
    const int b = vrbase >> 10;          // / (H*D) = 1024
    const int h = (vrbase >> 7) & 7;
    const int dbase = vrbase & 127;
    const int tid = threadIdx.x;
    const int cl = tid & 31;
    const int rl = tid >> 5;             // 0..7
    #pragma unroll
    for (int rr = 0; rr < 4; ++rr) {
        int t = tb * 32 + rl + rr * 8;
        tile[rl + rr * 8][cl] =
            qkv[(size_t)(b * T_DIM + t) * (3 * C_DIM) + 2 * C_DIM + h * D_DIM +
                dbase + cl];
    }
    __syncthreads();
    #pragma unroll
    for (int rr = 0; rr < 4; ++rr) {
        int vr = vrbase + rl + rr * 8;
        vT[(size_t)vr * T_DIM + tb * 32 + cl] = tile[cl][rl + rr * 8];
    }
}

// ---------------------------------------------------------------------------
// Flash-style causal attention with WMMA.  One wave handles 16 query rows of
// one (b,h).  qkv: [B*T, 3C] bf16; vT: [(B*H)*D, T] bf16; ybuf: [B*T, C].
// ---------------------------------------------------------------------------
__global__ __launch_bounds__(32)
void attn_kernel(const unsigned short* __restrict__ qkv,
                 const unsigned short* __restrict__ vT,
                 unsigned short* __restrict__ ybuf) {
    const int lane = threadIdx.x;
    const int idx  = blockIdx.x;
    const int qb   = idx & 63;        // T/16 = 64 query blocks
    const int h    = (idx >> 6) & 7;
    const int b    = idx >> 9;
    const int m0   = qb * 16;
    const int ln16 = lane & 15;
    const int half = lane >> 4;
    const int RS   = 3 * C_DIM;       // qkv row stride

    const unsigned short* qp = qkv + (size_t)b * T_DIM * RS + h * D_DIM;
    const unsigned short* kp = qp + C_DIM;
    const unsigned short* vTb = vT + (size_t)(b * H_DIM + h) * D_DIM * T_DIM;

    // Load q as 4 A-fragments (K = D = 128, chunks of 32)
    Frag qf[4];
    {
        const int base = half ? 8 : 0;
        const unsigned short* p = qp + (size_t)(m0 + ln16) * RS;
        #pragma unroll
        for (int kc = 0; kc < 4; ++kc) {
            qf[kc].q[0] = *reinterpret_cast<const uint4*>(p + kc * 32 + base);
            qf[kc].q[1] = *reinterpret_cast<const uint4*>(p + kc * 32 + base + 16);
        }
    }

    const v8f vz = {0.f, 0.f, 0.f, 0.f, 0.f, 0.f, 0.f, 0.f};
    v8f yacc[8];
    float mrow[8], lrow[8];
    #pragma unroll
    for (int r = 0; r < 8; ++r) {
        mrow[r] = -__builtin_inff();
        lrow[r] = 0.f;
        yacc[r] = vz;
    }

    __shared__ __align__(16) unsigned short Ps[16][40];

    const float scale = 0.0883883476483184f;  // 1/sqrt(128)
    const int kbB = half ? 8 : 0;             // B-fragment K offset (table order)

    for (int s0 = 0; s0 < m0 + 16; s0 += 32) {
        // ---- scores S = q @ k^T for a 16x32 block ----
        v8f sacc[2] = {vz, vz};
        #pragma unroll
        for (int j = 0; j < 2; ++j) {
            Frag kf[4];
            const unsigned short* p =
                kp + (size_t)(s0 + j * 16 + ln16) * RS + kbB;
            #pragma unroll
            for (int kc = 0; kc < 4; ++kc) {
                kf[kc].q[0] = *reinterpret_cast<const uint4*>(p + kc * 32);
                kf[kc].q[1] = *reinterpret_cast<const uint4*>(p + kc * 32 + 16);
            }
            #pragma unroll
            for (int kc = 0; kc < 4; ++kc)
                sacc[j] = __builtin_amdgcn_wmma_f32_16x16x32_bf16(
                    false, qf[kc].v, false, kf[kc].v, (short)0, sacc[j], false,
                    false);
        }

        // ---- online softmax update ----
        #pragma unroll
        for (int r = 0; r < 8; ++r) {
            const int row  = m0 + r + half * 8;
            const int col0 = s0 + ln16;
            const int col1 = col0 + 16;
            float x0 = sacc[0][r] * scale;
            float x1 = sacc[1][r] * scale;
            if (col0 > row) x0 = -__builtin_inff();
            if (col1 > row) x1 = -__builtin_inff();
            float mx = fmaxf(x0, x1);
            #pragma unroll
            for (int ml = 1; ml < 16; ml <<= 1)
                mx = fmaxf(mx, __shfl_xor(mx, ml, 32));
            float mnew = fmaxf(mrow[r], mx);
            float corr = __expf(mrow[r] - mnew);
            float e0 = __expf(x0 - mnew);
            float e1 = __expf(x1 - mnew);
            float sum = e0 + e1;
            #pragma unroll
            for (int ml = 1; ml < 16; ml <<= 1) sum += __shfl_xor(sum, ml, 32);
            lrow[r] = lrow[r] * corr + sum;
            mrow[r] = mnew;
            #pragma unroll
            for (int j2 = 0; j2 < 8; ++j2) yacc[j2][r] *= corr;
            Ps[r + half * 8][ln16]      = f2bf(e0);
            Ps[r + half * 8][ln16 + 16] = f2bf(e1);
        }
        __syncthreads();

        // ---- y += P (16x32) @ V (32x128), V fragments from vT rows ----
        Frag pf;
        {
            const int base = half ? 8 : 0;
            const unsigned short* pp = &Ps[ln16][0];
            pf.q[0] = *reinterpret_cast<const uint4*>(pp + base);
            pf.q[1] = *reinterpret_cast<const uint4*>(pp + base + 16);
        }
        #pragma unroll
        for (int j2 = 0; j2 < 8; ++j2) {
            Frag vf;
            const unsigned short* p =
                vTb + (size_t)(j2 * 16 + ln16) * T_DIM + s0 + kbB;
            vf.q[0] = *reinterpret_cast<const uint4*>(p);
            vf.q[1] = *reinterpret_cast<const uint4*>(p + 16);
            yacc[j2] = __builtin_amdgcn_wmma_f32_16x16x32_bf16(
                false, pf.v, false, vf.v, (short)0, yacc[j2], false, false);
        }
        __syncthreads();
    }

    // ---- normalize and write bf16 output ----
    #pragma unroll
    for (int j2 = 0; j2 < 8; ++j2) {
        #pragma unroll
        for (int r = 0; r < 8; ++r) {
            const int row = m0 + r + half * 8;
            float val = yacc[j2][r] / lrow[r];
            ybuf[(size_t)(b * T_DIM + row) * C_DIM + h * D_DIM + j2 * 16 + ln16] =
                f2bf(val);
        }
    }
}

// ---------------------------------------------------------------------------
// Orchestration
// ---------------------------------------------------------------------------
extern "C" void kernel_launch(void* const* d_in, const int* in_sizes, int n_in,
                              void* d_out, int out_size, void* d_ws, size_t ws_size,
                              hipStream_t stream) {
    const float* x      = (const float*)d_in[0];
    const float* wpe    = (const float*)d_in[1];
    const float* ln1_w  = (const float*)d_in[2];
    const float* ln1_b  = (const float*)d_in[3];
    const float* attn_w = (const float*)d_in[4];
    const float* attn_b = (const float*)d_in[5];
    const float* proj_w = (const float*)d_in[6];
    const float* proj_b = (const float*)d_in[7];
    const float* ln2_w  = (const float*)d_in[8];
    const float* ln2_b  = (const float*)d_in[9];
    const float* fc_w   = (const float*)d_in[10];
    const float* fc_b   = (const float*)d_in[11];
    const float* fcp_w  = (const float*)d_in[12];
    const float* fcp_b  = (const float*)d_in[13];
    const float* lnf_w  = (const float*)d_in[14];
    const float* lnf_b  = (const float*)d_in[15];

    char* ws = (char*)d_ws;
    size_t off = 0;
    auto alloc = [&](size_t bytes) -> char* {
        char* p = ws + off;
        off += (bytes + 255) & ~(size_t)255;
        return p;
    };

    const size_t nAttnW = (size_t)L_DIM * 3 * C_DIM * C_DIM;
    const size_t nProjW = (size_t)L_DIM * C_DIM * C_DIM;
    const size_t nFcW   = (size_t)L_DIM * 4 * C_DIM * C_DIM;
    const size_t nFcpW  = (size_t)L_DIM * C_DIM * 4 * C_DIM;

    unsigned short* wAttn = (unsigned short*)alloc(nAttnW * 2);
    unsigned short* wProj = (unsigned short*)alloc(nProjW * 2);
    unsigned short* wFc   = (unsigned short*)alloc(nFcW * 2);
    unsigned short* wFcp  = (unsigned short*)alloc(nFcpW * 2);
    float*          h     = (float*)alloc((size_t)M_ROWS * C_DIM * 4);
    unsigned short* xn    = (unsigned short*)alloc((size_t)M_ROWS * C_DIM * 2);
    unsigned short* qkv   = (unsigned short*)alloc((size_t)M_ROWS * 3 * C_DIM * 2);
    unsigned short* vT    = (unsigned short*)alloc((size_t)B_DIM * H_DIM * D_DIM * T_DIM * 2);
    unsigned short* ybuf  = (unsigned short*)alloc((size_t)M_ROWS * C_DIM * 2);
    unsigned short* act4  = (unsigned short*)alloc((size_t)M_ROWS * 4 * C_DIM * 2);

    f32_to_bf16_kernel<<<4096, 256, 0, stream>>>(attn_w, wAttn, nAttnW);
    f32_to_bf16_kernel<<<4096, 256, 0, stream>>>(proj_w, wProj, nProjW);
    f32_to_bf16_kernel<<<4096, 256, 0, stream>>>(fc_w,   wFc,   nFcW);
    f32_to_bf16_kernel<<<4096, 256, 0, stream>>>(fcp_w,  wFcp,  nFcpW);

    embed_kernel<<<(M_ROWS * C_DIM) / 256, 256, 0, stream>>>(x, wpe, h);

    for (int l = 0; l < L_DIM; ++l) {
        ln_kernel<true><<<M_ROWS, 256, 0, stream>>>(
            h, ln1_w + l * C_DIM, ln1_b + l * C_DIM, xn);
        gemm_bf16<0><<<dim3(3 * C_DIM / 128, M_ROWS / 128), 256, 0, stream>>>(
            xn, wAttn + (size_t)l * 3 * C_DIM * C_DIM, attn_b + l * 3 * C_DIM,
            qkv, nullptr, M_ROWS, 3 * C_DIM, C_DIM);
        v_transpose_kernel<<<dim3(T_DIM / 32, (B_DIM * H_DIM * D_DIM) / 32),
                             256, 0, stream>>>(qkv, vT);
        attn_kernel<<<B_DIM * H_DIM * (T_DIM / 16), 32, 0, stream>>>(qkv, vT, ybuf);
        gemm_bf16<2><<<dim3(C_DIM / 128, M_ROWS / 128), 256, 0, stream>>>(
            ybuf, wProj + (size_t)l * C_DIM * C_DIM, proj_b + l * C_DIM,
            h, h, M_ROWS, C_DIM, C_DIM);
        ln_kernel<true><<<M_ROWS, 256, 0, stream>>>(
            h, ln2_w + l * C_DIM, ln2_b + l * C_DIM, xn);
        gemm_bf16<1><<<dim3(4 * C_DIM / 128, M_ROWS / 128), 256, 0, stream>>>(
            xn, wFc + (size_t)l * 4 * C_DIM * C_DIM, fc_b + l * 4 * C_DIM,
            act4, nullptr, M_ROWS, 4 * C_DIM, C_DIM);
        gemm_bf16<2><<<dim3(C_DIM / 128, M_ROWS / 128), 256, 0, stream>>>(
            act4, wFcp + (size_t)l * C_DIM * 4 * C_DIM, fcp_b + l * C_DIM,
            h, h, M_ROWS, C_DIM, 4 * C_DIM);
    }

    ln_kernel<false><<<M_ROWS, 256, 0, stream>>>(h, lnf_w, lnf_b, (float*)d_out);

    (void)in_sizes; (void)n_in; (void)out_size; (void)ws_size;
}
